// BurgersPINN_1288490188827
// MI455X (gfx1250) — compile-verified, matrix operand order checked by reading, not run on hardware
//
#include <hip/hip_runtime.h>
#include <hip/hip_bf16.h>

typedef __attribute__((ext_vector_type(16))) _Float16     v16h;
typedef __attribute__((ext_vector_type(8)))  float        v8f;
typedef __attribute__((ext_vector_type(8)))  unsigned int v8u;

#define HID    20
#define KPAD   32
#define NWAVES 8
#define CWGS   512
#define DWGS   128

static __device__ const float NU_C = 0.0031830988618367176f; // 0.01/pi

// ---------- helpers ----------

// Branch-free tanh from native exp2/rcp:
// tanh(x) = 1 - 2/(1 + 2^(2*log2(e)*x)); saturates correctly at +/-inf.
__device__ __forceinline__ float fast_tanh(float x) {
  float ex = __builtin_amdgcn_exp2f(x * 2.8853900817779268f); // v_exp_f32
  float r  = __builtin_amdgcn_rcpf(ex + 1.0f);                // v_rcp_f32
  return 1.0f - (r + r);
}

__device__ __forceinline__ unsigned packB(const float* W, int K0, int n) {
  // W is [fan_in=20][fan_out=20] row-major; pack two consecutive-K f16 values
  float a = (K0     < HID && n < HID) ? W[K0 * HID + n]       : 0.f;
  float b = (K0 + 1 < HID && n < HID) ? W[(K0 + 1) * HID + n] : 0.f;
  unsigned short ua = __builtin_bit_cast(unsigned short, (_Float16)a);
  unsigned short ub = __builtin_bit_cast(unsigned short, (_Float16)b);
  return (unsigned)ua | ((unsigned)ub << 16);
}

__device__ __forceinline__ void preload_weights(
    float* sW1, float* sB1, float* sB2, float* sB3, float* sW4, float* sB4,
    unsigned* sW2B, unsigned* sW3B,
    const float* W1, const float* b1, const float* W2, const float* b2,
    const float* W3, const float* b3, const float* W4, const float* b4,
    int tid, int nthr)
{
  for (int i = tid; i < 40; i += nthr) sW1[i] = W1[i];
  for (int i = tid; i < HID; i += nthr) { sB1[i] = b1[i]; sW4[i] = W4[i]; }
  for (int i = tid; i < 32; i += nthr) {
    sB2[i] = (i < HID) ? b2[i] : 0.f;
    sB3[i] = (i < HID) ? b3[i] : 0.f;
  }
  // Pre-pack W2/W3 into per-lane B-fragment register order:
  // index = nt*256 + lane*8 + r ; VGPR r holds K pair per 16-bit A/B layout
  for (int i = tid; i < 512; i += nthr) {
    int nt = i >> 8, rem = i & 255, lane = rem >> 3, r = rem & 7;
    int n  = nt * 16 + (lane & 15);
    int kh = lane >> 4;
    int K0 = ((r < 4) ? 2 * r : 16 + 2 * (r - 4)) + kh * 8;
    sW2B[i] = packB(W2, K0, n);
    sW3B[i] = packB(W3, K0, n);
  }
  if (tid == 0) sB4[0] = b4[0];
}

// A fragment: 16x32 f16, row = one point (64B). lanes0-15: K0-7 & K16-23,
// lanes16-31: K8-15 & K24-31 -> two 16B loads per lane.
__device__ __forceinline__ v16h loadA(const _Float16* row, int kh) {
  const uint4* p = (const uint4*)row;
  uint4 lo = p[kh];
  uint4 hi = p[2 + kh];
  v8u u;
  u[0] = lo.x; u[1] = lo.y; u[2] = lo.z; u[3] = lo.w;
  u[4] = hi.x; u[5] = hi.y; u[6] = hi.z; u[7] = hi.w;
  return __builtin_bit_cast(v16h, u);
}

__device__ __forceinline__ v16h loadBfrag(const unsigned* base, int nt, int lane) {
  const uint4* p = (const uint4*)(base + nt * 256 + lane * 8);
  uint4 lo = p[0], hi = p[1];
  v8u u;
  u[0] = lo.x; u[1] = lo.y; u[2] = lo.z; u[3] = lo.w;
  u[4] = hi.x; u[5] = hi.y; u[6] = hi.z; u[7] = hi.w;
  return __builtin_bit_cast(v16h, u);
}

__device__ __forceinline__ v8f wmma16(v16h a, v16h b, v8f c) {
  return __builtin_amdgcn_wmma_f32_16x16x32_f16(
      /*neg_a=*/false, a, /*neg_b=*/false, b,
      /*c_mod=*/(short)0, c, /*reuse_a=*/false, /*reuse_b=*/false);
}

__device__ __forceinline__ v8f splat8(float x) {
  v8f v;
#pragma unroll
  for (int j = 0; j < 8; j++) v[j] = x;
  return v;
}

__device__ __forceinline__ float2 unp2(unsigned d) {
  float lo = (float)__builtin_bit_cast(_Float16, (unsigned short)(d & 0xffffu));
  float hi = (float)__builtin_bit_cast(_Float16, (unsigned short)(d >> 16));
  return make_float2(lo, hi);
}

// ---------- collocation kernel: 4-channel jet (u, u_t, u_x, u_xx) ----------

__global__ __launch_bounds__(256) void burgers_colloc(
    const float* __restrict__ tc, const float* __restrict__ xc,
    const float* __restrict__ W1, const float* __restrict__ b1,
    const float* __restrict__ W2, const float* __restrict__ b2,
    const float* __restrict__ W3, const float* __restrict__ b3,
    const float* __restrict__ W4, const float* __restrict__ b4,
    int ntiles, float* __restrict__ partial)
{
  __shared__ float sW1[40], sB1[HID], sB2[32], sB3[32], sW4[HID], sB4[1];
  __shared__ alignas(16) unsigned sW2B[512], sW3B[512];
  __shared__ alignas(64) _Float16 sAct[NWAVES][4 * 16 * KPAD]; // [ch][p][k]
  __shared__ float sTX[NWAVES][32];
  __shared__ float sRed[NWAVES];

  const int tid = threadIdx.x;
  preload_weights(sW1, sB1, sB2, sB3, sW4, sB4, sW2B, sW3B,
                  W1, b1, W2, b2, W3, b3, W4, b4, tid, blockDim.x);
  __syncthreads();

  const int lane = tid & 31, wv = tid >> 5;
  const int wavesTotal = gridDim.x * (blockDim.x >> 5);
  const int gw = blockIdx.x * (blockDim.x >> 5) + wv;

  v16h B2f0 = loadBfrag(sW2B, 0, lane), B2f1 = loadBfrag(sW2B, 1, lane);
  v16h B3f0 = loadBfrag(sW3B, 0, lane), B3f1 = loadBfrag(sW3B, 1, lane);
  const int n0 = lane & 15, n1 = 16 + (lane & 15);
  v8f cb20 = splat8(sB2[n0]), cb21 = splat8(sB2[n1]);
  v8f cb30 = splat8(sB3[n0]), cb31 = splat8(sB3[n1]);
  v8f zf = splat8(0.f);
  const float b4v = sB4[0];

  _Float16* act = sAct[wv];
  float* txs = sTX[wv];
  float acc = 0.f;
  const int kh = lane >> 4;
  const int arow = lane & 15;
  const int prow = kh << 3;

  for (int tile = gw; tile < ntiles; tile += wavesTotal) {
    if (lane < 16) {
      int p = tile * 16 + lane;
      txs[lane]      = tc[p];
      txs[16 + lane] = xc[p];
    }
    // ---- layer 1 (K=2): plain VALU, lane = neuron ----
    {
      int n = lane;
      bool vn = n < HID;
      int nc = vn ? n : 0;
      float w0 = vn ? sW1[nc] : 0.f;
      float w1 = vn ? sW1[20 + nc] : 0.f;
      float bb = vn ? sB1[nc] : 0.f;
      for (int p = 0; p < 16; p++) {
        float z = txs[p] * w0 + txs[16 + p] * w1 + bb;
        float v = fast_tanh(z);
        float s = 1.f - v * v;
        int base = p * KPAD + n;
        act[0 * 512 + base] = (_Float16)(vn ? v : 0.f);
        act[1 * 512 + base] = (_Float16)(s * w0);
        act[2 * 512 + base] = (_Float16)(s * w1);
        act[3 * 512 + base] = (_Float16)(-2.f * v * s * w1 * w1);
      }
    }
    // ---- layers 2 & 3: WMMA, 4 channels x 2 N-tiles ----
#pragma unroll
    for (int L = 0; L < 2; L++) {
      v16h aV  = loadA(act + 0 * 512 + arow * KPAD, kh);
      v16h aT  = loadA(act + 1 * 512 + arow * KPAD, kh);
      v16h aX  = loadA(act + 2 * 512 + arow * KPAD, kh);
      v16h aXX = loadA(act + 3 * 512 + arow * KPAD, kh);
      v16h bf0 = L ? B3f0 : B2f0;
      v16h bf1 = L ? B3f1 : B2f1;
      v8f zV0  = wmma16(aV,  bf0, L ? cb30 : cb20);
      v8f zV1  = wmma16(aV,  bf1, L ? cb31 : cb21);
      v8f zT0  = wmma16(aT,  bf0, zf);
      v8f zT1  = wmma16(aT,  bf1, zf);
      v8f zX0  = wmma16(aX,  bf0, zf);
      v8f zX1  = wmma16(aX,  bf1, zf);
      v8f zXX0 = wmma16(aXX, bf0, zf);
      v8f zXX1 = wmma16(aXX, bf1, zf);
#pragma unroll
      for (int r = 0; r < 8; r++) {
        int p = prow + r;
        {
          float v = fast_tanh(zV0[r]); float s = 1.f - v * v; float dx = zX0[r];
          int base = p * KPAD + n0;
          act[0 * 512 + base] = (_Float16)v;
          act[1 * 512 + base] = (_Float16)(s * zT0[r]);
          act[2 * 512 + base] = (_Float16)(s * dx);
          act[3 * 512 + base] = (_Float16)(s * zXX0[r] - 2.f * v * s * dx * dx);
        }
        {
          float v = fast_tanh(zV1[r]); float s = 1.f - v * v; float dx = zX1[r];
          int base = p * KPAD + n1;
          act[0 * 512 + base] = (_Float16)v;
          act[1 * 512 + base] = (_Float16)(s * zT1[r]);
          act[2 * 512 + base] = (_Float16)(s * dx);
          act[3 * 512 + base] = (_Float16)(s * zXX1[r] - 2.f * v * s * dx * dx);
        }
      }
    }
    // ---- layer 4 (N=1): per-point dot, residual ----
    if (lane < 16) {
      int p = lane;
      float u = b4v, ut = 0.f, ux = 0.f, uxx = 0.f;
      const unsigned* rV  = (const unsigned*)(act + 0 * 512 + p * KPAD);
      const unsigned* rT  = (const unsigned*)(act + 1 * 512 + p * KPAD);
      const unsigned* rX  = (const unsigned*)(act + 2 * 512 + p * KPAD);
      const unsigned* rXX = (const unsigned*)(act + 3 * 512 + p * KPAD);
#pragma unroll
      for (int n2 = 0; n2 < 10; n2++) {
        float w0 = sW4[2 * n2], w1 = sW4[2 * n2 + 1];
        float2 aV  = unp2(rV[n2]);
        float2 aT  = unp2(rT[n2]);
        float2 aX  = unp2(rX[n2]);
        float2 aXX = unp2(rXX[n2]);
        u   += aV.x * w0 + aV.y * w1;
        ut  += aT.x * w0 + aT.y * w1;
        ux  += aX.x * w0 + aX.y * w1;
        uxx += aXX.x * w0 + aXX.y * w1;
      }
      float f = ut + u * ux - NU_C * uxx;
      acc += f * f;
    }
  }

#pragma unroll
  for (int off = 16; off > 0; off >>= 1) acc += __shfl_xor(acc, off, 32);
  if (lane == 0) sRed[wv] = acc;
  __syncthreads();
  if (tid == 0) {
    float s = 0.f;
    for (int w = 0; w < NWAVES; w++) s += sRed[w];
    partial[blockIdx.x] = s;
  }
}

// ---------- data-fit kernel: 1-channel forward, (u - u_true)^2 ----------

__global__ __launch_bounds__(256) void burgers_fit(
    const float* __restrict__ tp, const float* __restrict__ xp,
    const float* __restrict__ utrue,
    const float* __restrict__ W1, const float* __restrict__ b1,
    const float* __restrict__ W2, const float* __restrict__ b2,
    const float* __restrict__ W3, const float* __restrict__ b3,
    const float* __restrict__ W4, const float* __restrict__ b4,
    int ntiles, float* __restrict__ partial)
{
  __shared__ float sW1[40], sB1[HID], sB2[32], sB3[32], sW4[HID], sB4[1];
  __shared__ alignas(16) unsigned sW2B[512], sW3B[512];
  __shared__ alignas(64) _Float16 sAct[NWAVES][16 * KPAD];
  __shared__ float sTX[NWAVES][32];
  __shared__ float sRed[NWAVES];

  const int tid = threadIdx.x;
  preload_weights(sW1, sB1, sB2, sB3, sW4, sB4, sW2B, sW3B,
                  W1, b1, W2, b2, W3, b3, W4, b4, tid, blockDim.x);
  __syncthreads();

  const int lane = tid & 31, wv = tid >> 5;
  const int wavesTotal = gridDim.x * (blockDim.x >> 5);
  const int gw = blockIdx.x * (blockDim.x >> 5) + wv;

  v16h B2f0 = loadBfrag(sW2B, 0, lane), B2f1 = loadBfrag(sW2B, 1, lane);
  v16h B3f0 = loadBfrag(sW3B, 0, lane), B3f1 = loadBfrag(sW3B, 1, lane);
  const int n0 = lane & 15, n1 = 16 + (lane & 15);
  v8f cb20 = splat8(sB2[n0]), cb21 = splat8(sB2[n1]);
  v8f cb30 = splat8(sB3[n0]), cb31 = splat8(sB3[n1]);
  const float b4v = sB4[0];

  _Float16* act = sAct[wv];
  float* txs = sTX[wv];
  float acc = 0.f;
  const int kh = lane >> 4;
  const int arow = lane & 15;
  const int prow = kh << 3;

  for (int tile = gw; tile < ntiles; tile += wavesTotal) {
    if (lane < 16) {
      int p = tile * 16 + lane;
      txs[lane]      = tp[p];
      txs[16 + lane] = xp[p];
    }
    // layer 1
    {
      int n = lane;
      bool vn = n < HID;
      int nc = vn ? n : 0;
      float w0 = vn ? sW1[nc] : 0.f;
      float w1 = vn ? sW1[20 + nc] : 0.f;
      float bb = vn ? sB1[nc] : 0.f;
      for (int p = 0; p < 16; p++) {
        float z = txs[p] * w0 + txs[16 + p] * w1 + bb;
        float v = fast_tanh(z);
        act[p * KPAD + n] = (_Float16)(vn ? v : 0.f);
      }
    }
    // layers 2 & 3
#pragma unroll
    for (int L = 0; L < 2; L++) {
      v16h aV  = loadA(act + arow * KPAD, kh);
      v16h bf0 = L ? B3f0 : B2f0;
      v16h bf1 = L ? B3f1 : B2f1;
      v8f zV0 = wmma16(aV, bf0, L ? cb30 : cb20);
      v8f zV1 = wmma16(aV, bf1, L ? cb31 : cb21);
#pragma unroll
      for (int r = 0; r < 8; r++) {
        int p = prow + r;
        act[p * KPAD + n0] = (_Float16)fast_tanh(zV0[r]);
        act[p * KPAD + n1] = (_Float16)fast_tanh(zV1[r]);
      }
    }
    // layer 4 + loss
    if (lane < 16) {
      int p = lane;
      float u = b4v;
      const unsigned* rV = (const unsigned*)(act + p * KPAD);
#pragma unroll
      for (int n2 = 0; n2 < 10; n2++) {
        float2 aV = unp2(rV[n2]);
        u += aV.x * sW4[2 * n2] + aV.y * sW4[2 * n2 + 1];
      }
      float d = u - utrue[tile * 16 + p];
      acc += d * d;
    }
  }

#pragma unroll
  for (int off = 16; off > 0; off >>= 1) acc += __shfl_xor(acc, off, 32);
  if (lane == 0) sRed[wv] = acc;
  __syncthreads();
  if (tid == 0) {
    float s = 0.f;
    for (int w = 0; w < NWAVES; w++) s += sRed[w];
    partial[blockIdx.x] = s;
  }
}

// ---------- deterministic finalize ----------

__global__ void burgers_final(const float* __restrict__ part, float* __restrict__ out,
                              float nC, float nI, float nB)
{
  if (threadIdx.x == 0 && blockIdx.x == 0) {
    float sF = 0.f, sI = 0.f, sB = 0.f;
    for (int i = 0; i < CWGS; i++) sF += part[i];
    for (int i = 0; i < DWGS; i++) sI += part[CWGS + i];
    for (int i = 0; i < DWGS; i++) sB += part[CWGS + DWGS + i];
    out[0] = sF / nC + sI / nI + sB / nB;
  }
}

// ---------- host launcher ----------

extern "C" void kernel_launch(void* const* d_in, const int* in_sizes, int n_in,
                              void* d_out, int out_size, void* d_ws, size_t ws_size,
                              hipStream_t stream)
{
  const float* tc = (const float*)d_in[0];
  const float* xc = (const float*)d_in[1];
  const float* ti = (const float*)d_in[2];
  const float* xi = (const float*)d_in[3];
  const float* ui = (const float*)d_in[4];
  const float* tb = (const float*)d_in[5];
  const float* xb = (const float*)d_in[6];
  const float* ub = (const float*)d_in[7];
  const float* W1 = (const float*)d_in[8];
  const float* b1 = (const float*)d_in[9];
  const float* W2 = (const float*)d_in[10];
  const float* b2 = (const float*)d_in[11];
  const float* W3 = (const float*)d_in[12];
  const float* b3 = (const float*)d_in[13];
  const float* W4 = (const float*)d_in[14];
  const float* b4 = (const float*)d_in[15];

  float* part = (float*)d_ws;
  int ntC = in_sizes[0] / 16;
  int ntI = in_sizes[2] / 16;
  int ntB = in_sizes[5] / 16;

  burgers_colloc<<<CWGS, 256, 0, stream>>>(tc, xc, W1, b1, W2, b2, W3, b3, W4, b4,
                                           ntC, part);
  burgers_fit<<<DWGS, 256, 0, stream>>>(ti, xi, ui, W1, b1, W2, b2, W3, b3, W4, b4,
                                        ntI, part + CWGS);
  burgers_fit<<<DWGS, 256, 0, stream>>>(tb, xb, ub, W1, b1, W2, b2, W3, b3, W4, b4,
                                        ntB, part + CWGS + DWGS);
  burgers_final<<<1, 1, 0, stream>>>(part, (float*)d_out,
                                     (float)in_sizes[0], (float)in_sizes[2],
                                     (float)in_sizes[5]);
}
